// SynthesisNetwork_38689065403010
// MI455X (gfx1250) — matrix operand order repaired and checked by reading, not compiled, not run on hardware
//
#include <hip/hip_runtime.h>
#include <hip/hip_bf16.h>

// ---------- types for WMMA ----------
typedef __attribute__((ext_vector_type(16))) __bf16 v16bf;
typedef __attribute__((ext_vector_type(8)))  float  v8f;
union FragBF { v16bf v; unsigned u[8]; };

// ---------- sizes ----------
#define BATCH 4096
#define HID   400
#define NWIN  64
#define AWIN  80
#define NGAUSS 10
#define NMIX  20
#define NCOL  1600   // 4*HID

// output float offsets (flat, reference return order)
#define OFF_PI   0
#define OFF_MU   20
#define OFF_SD   60
#define OFF_RO   100
#define OFF_EOS  120
#define OFF_W    121
#define OFF_K    327801
#define OFF_H1   368761
#define OFF_C1   2007161
#define OFF_H2   3645561
#define OFF_C2   5283961
#define OFF_H3   6922361
#define OFF_C3   8560761
#define OFF_PHI  10199161

__device__ __forceinline__ unsigned short f32_to_bf16u(float f) {
  unsigned u = __builtin_bit_cast(unsigned, f);
  unsigned r = u + 0x7FFFu + ((u >> 16) & 1u);   // round-to-nearest-even
  return (unsigned short)(r >> 16);
}
__device__ __forceinline__ float sigmoidf(float x) { return 1.f / (1.f + __expf(-x)); }

// ---------- W_xh f32 -> bf16, TRANSPOSED: dst[n*kpad + k], K zero-padded ----------
__global__ __launch_bounds__(256) void convert_w_t(const float* __restrict__ src,
                                                   unsigned short* __restrict__ dst,
                                                   int K, int kpad) {
  __shared__ unsigned short tile[32][34];      // padded to dodge bank conflicts
  const int kb = blockIdx.x * 32, nb = blockIdx.y * 32;
  const int tx = threadIdx.x & 31, ty = threadIdx.x >> 5;   // 32 x 8
  #pragma unroll
  for (int r = 0; r < 32; r += 8) {
    const int kk = kb + ty + r;
    const float v = (kk < K) ? src[(size_t)kk * NCOL + (nb + tx)] : 0.f;
    tile[ty + r][tx] = f32_to_bf16u(v);
  }
  __syncthreads();
  #pragma unroll
  for (int r = 0; r < 32; r += 8) {
    const int n = nb + ty + r;
    dst[(size_t)n * kpad + (kb + tx)] = tile[tx][ty + r];
  }
}

// ---------- gather concatenated LSTM input [x(3)|p1(l1)|p2(l2)|p3(l3)] -> bf16 ----------
__global__ void gather_a(unsigned short* __restrict__ dst, int kpad,
                         const float* __restrict__ p0, const float* __restrict__ p1,
                         const float* __restrict__ p2, const float* __restrict__ p3,
                         int l1, int l2, int l3) {
  int idx = blockIdx.x * 256 + threadIdx.x;
  if (idx >= BATCH * kpad) return;
  int b = idx / kpad, col = idx % kpad;
  float v = 0.f;
  if (col < 3) v = p0[b * 3 + col];
  else {
    int c = col - 3;
    if (c < l1) v = p1[b * l1 + c];
    else { c -= l1;
      if (c < l2) v = p2[b * l2 + c];
      else { c -= l2;
        if (c < l3) v = p3[b * l3 + c];
      }
    }
  }
  dst[idx] = f32_to_bf16u(v);
}

// ---------- bf16 WMMA GEMM: Z[B x 1600] = A[B x kpad] * WT^T (WT is [1600 x kpad]) ----------
__global__ __launch_bounds__(128) void gemm_bf16_wmma(
    const unsigned short* __restrict__ A, const unsigned short* __restrict__ WT,
    float* __restrict__ Z, int kpad) {
  __shared__ __align__(16) unsigned short As[64 * 32];   // [row][k]
  __shared__ __align__(16) unsigned short WsT[64 * 32];  // [col][k]
  const int tid  = threadIdx.x;
  const int lane = tid & 31;
  const int wave = tid >> 5;
  const int wy = wave >> 1, wx = wave & 1;      // 2x2 waves -> 64x64 block tile
  const int m0 = blockIdx.x * 64;
  const int n0 = blockIdx.y * 64;
  const int hf  = lane >> 4;                    // lane half (ISA 16-bit layouts)
  const int l15 = lane & 15;

  // staging decomposition (identical for A rows and WT cols): 4 x 8-bf16 chunks
  const int s_ch = tid & 3;    // 16B chunk within the 32-k row
  const int s_r0 = tid >> 2;   // 32 rows per pass, 2 passes

  v8f acc[2][2];
  #pragma unroll
  for (int i = 0; i < 2; ++i)
    #pragma unroll
    for (int j = 0; j < 2; ++j)
      #pragma unroll
      for (int e = 0; e < 8; ++e) acc[i][j][e] = 0.f;

  const int nIter = kpad >> 5;
  for (int kt = 0; kt < nIter; ++kt) {
    #pragma unroll
    for (int p = 0; p < 2; ++p) {
      const int r = s_r0 + p * 32;
      *(uint4*)&As[r * 32 + s_ch * 8] =
          *(const uint4*)(A + (size_t)(m0 + r) * kpad + (size_t)kt * 32 + s_ch * 8);
      *(uint4*)&WsT[r * 32 + s_ch * 8] =
          *(const uint4*)(WT + (size_t)(n0 + r) * kpad + (size_t)kt * 32 + s_ch * 8);
    }
    if (kt + 1 < nIter) {  // prefetch next k-tile
      __builtin_prefetch(A  + (size_t)(m0 + s_r0) * kpad + (size_t)(kt + 1) * 32 + s_ch * 8, 0, 3);
      __builtin_prefetch(WT + (size_t)(n0 + s_r0) * kpad + (size_t)(kt + 1) * 32 + s_ch * 8, 0, 3);
    }
    __syncthreads();

    FragBF af[2], bw[2];
    #pragma unroll
    for (int i = 0; i < 2; ++i) {
      // A 16x32: lanes 0-15 row M=l15 K0..15, lanes 16-31 K16..31
      const unsigned* s = (const unsigned*)&As[(wy * 32 + i * 16 + l15) * 32];
      const int base = 8 * hf;
      #pragma unroll
      for (int t = 0; t < 8; ++t) af[i].u[t] = s[base + t];
    }
    #pragma unroll
    for (int j = 0; j < 2; ++j) {
      // B 32x16: lanes 0-15 col N=l15 K0..15, lanes 16-31 K16..31
      const unsigned* s = (const unsigned*)&WsT[(wx * 32 + j * 16 + l15) * 32];
      const int base = 8 * hf;
      #pragma unroll
      for (int t = 0; t < 8; ++t) bw[j].u[t] = s[base + t];
    }
    #pragma unroll
    for (int i = 0; i < 2; ++i)
      #pragma unroll
      for (int j = 0; j < 2; ++j)
        acc[i][j] = __builtin_amdgcn_wmma_f32_16x16x32_bf16(
            false, af[i].v, false, bw[j].v, (short)0, acc[i][j], false, false);
    __syncthreads();
  }

  // C/D layout: VGPR r, lane -> row = r + 8*hf, col = l15
  #pragma unroll
  for (int i = 0; i < 2; ++i) {
    const int gr = m0 + wy * 32 + i * 16 + 8 * hf;
    #pragma unroll
    for (int j = 0; j < 2; ++j) {
      const int gc = n0 + wx * 32 + j * 16 + l15;
      #pragma unroll
      for (int r = 0; r < 8; ++r)
        Z[(size_t)(gr + r) * NCOL + gc] = acc[i][j][r];
    }
  }
}

// ---------- LSTM pointwise cell with peephole ----------
__global__ void lstm_cell(const float* __restrict__ Z, const float* __restrict__ cprev,
                          const float* __restrict__ Wci, const float* __restrict__ bi,
                          const float* __restrict__ Wcf, const float* __restrict__ bfv,
                          const float* __restrict__ bc,  const float* __restrict__ Wco,
                          const float* __restrict__ bo,
                          float* __restrict__ hout, float* __restrict__ cout) {
  int idx = blockIdx.x * blockDim.x + threadIdx.x;
  if (idx >= BATCH * HID) return;
  int b = idx / HID, j = idx % HID;
  const float* z = Z + (size_t)b * NCOL;
  float cp = cprev[idx];
  float it = sigmoidf(z[j]            + cp * Wci[j] + bi[j]);
  float ft = sigmoidf(z[HID + j]      + cp * Wcf[j] + bfv[j]);
  float cn = ft * cp + it * tanhf(z[2 * HID + j] + bc[j]);
  float ot = sigmoidf(z[3 * HID + j]  + cn * Wco[j] + bo[j]);
  hout[idx] = ot * tanhf(cn);
  cout[idx] = cn;
}

// ---------- attention window: alpha/beta/kappa + phi ----------
__global__ __launch_bounds__(128) void window_k(
    const float* __restrict__ h1n,
    const float* __restrict__ Wa, const float* __restrict__ ba,
    const float* __restrict__ Wb, const float* __restrict__ bb,
    const float* __restrict__ Wk, const float* __restrict__ bk,
    const float* __restrict__ kin, float* __restrict__ knew_out,
    float* __restrict__ phi_out) {
  __shared__ float hs[HID];
  __shared__ float ag[NGAUSS], bg[NGAUSS], kg[NGAUSS];
  int b = blockIdx.x, t = threadIdx.x;
  for (int i = t; i < HID; i += 128) hs[i] = h1n[(size_t)b * HID + i];
  __syncthreads();
  if (t < 3 * NGAUSS) {
    int g = t % NGAUSS, head = t / NGAUSS;
    const float* W  = (head == 0) ? Wa : (head == 1) ? Wb : Wk;
    const float* bv = (head == 0) ? ba : (head == 1) ? bb : bk;
    float d = bv[g];
    for (int h = 0; h < HID; ++h) d += hs[h] * W[h * NGAUSS + g];
    float e = __expf(d);
    if (head == 0) ag[g] = e;
    else if (head == 1) bg[g] = e;
    else { float kn = kin[b * NGAUSS + g] + e; kg[g] = kn; knew_out[b * NGAUSS + g] = kn; }
  }
  __syncthreads();
  if (t < NWIN) {
    float u = (float)t, p = 0.f;
    #pragma unroll
    for (int g = 0; g < NGAUSS; ++g) { float dk = kg[g] - u; p += ag[g] * __expf(-bg[g] * dk * dk); }
    phi_out[b * NWIN + t] = p;
  }
}

// ---------- w_new = phi @ c ----------
__global__ __launch_bounds__(128) void wnew_k(const float* __restrict__ phi,
                                              const float* __restrict__ c,
                                              float* __restrict__ wout) {
  __shared__ float ps[NWIN];
  int b = blockIdx.x, t = threadIdx.x;
  if (t < NWIN) ps[t] = phi[b * NWIN + t];
  __syncthreads();
  if (t < AWIN) {
    const float* cb = c + (size_t)b * (NWIN * AWIN) + t;
    float acc = 0.f;
    #pragma unroll 8
    for (int n = 0; n < NWIN; ++n) acc += ps[n] * cb[n * AWIN];
    wout[b * AWIN + t] = acc;
  }
}

// ---------- mixture heads for batch row 0 only ----------
__global__ __launch_bounds__(256) void mix_head(
    const float* __restrict__ h1, const float* __restrict__ h2, const float* __restrict__ h3,
    const float* __restrict__ Wpi, const float* __restrict__ bpi,
    const float* __restrict__ Wmu, const float* __restrict__ bmu,
    const float* __restrict__ Wsd, const float* __restrict__ bsd,
    const float* __restrict__ Wro, const float* __restrict__ bro,
    const float* __restrict__ Weo, const float* __restrict__ beo,
    const float* __restrict__ bias, float* __restrict__ out) {
  __shared__ float hc[3 * HID];
  __shared__ float logits[NMIX];
  int t = threadIdx.x;
  for (int i = t; i < HID; i += 256) { hc[i] = h1[i]; hc[HID + i] = h2[i]; hc[2 * HID + i] = h3[i]; }
  __syncthreads();
  float b0 = bias[0];
  if (t < 20) {                 // pi logits
    float d = bpi[t];
    for (int h = 0; h < 3 * HID; ++h) d += hc[h] * Wpi[h * 20 + t];
    logits[t] = d * (1.f + b0);
  } else if (t < 60) {          // mu
    int o = t - 20; float d = bmu[o];
    for (int h = 0; h < 3 * HID; ++h) d += hc[h] * Wmu[h * 40 + o];
    out[OFF_MU + o] = d;
  } else if (t < 100) {         // sd
    int o = t - 60; float d = bsd[o];
    for (int h = 0; h < 3 * HID; ++h) d += hc[h] * Wsd[h * 40 + o];
    out[OFF_SD + o] = __expf(d - b0);
  } else if (t < 120) {         // ro
    int o = t - 100; float d = bro[o];
    for (int h = 0; h < 3 * HID; ++h) d += hc[h] * Wro[h * 20 + o];
    out[OFF_RO + o] = tanhf(d);
  } else if (t == 120) {        // eos
    float d = beo[0];
    for (int h = 0; h < 3 * HID; ++h) d += hc[h] * Weo[h];
    out[OFF_EOS] = sigmoidf(d);
  }
  __syncthreads();
  if (t < 20) {
    float mx = -1e30f;
    for (int i = 0; i < 20; ++i) mx = fmaxf(mx, logits[i]);
    float s = 0.f;
    for (int i = 0; i < 20; ++i) s += __expf(logits[i] - mx);
    out[OFF_PI + t] = __expf(logits[t] - mx) / s;
  }
}

extern "C" void kernel_launch(void* const* d_in, const int* in_sizes, int n_in,
                              void* d_out, int out_size, void* d_ws, size_t ws_size,
                              hipStream_t stream) {
  (void)in_sizes; (void)n_in; (void)out_size; (void)ws_size;
  const float* x    = (const float*)d_in[0];
  const float* cmat = (const float*)d_in[1];
  const float* w0   = (const float*)d_in[2];
  const float* kin  = (const float*)d_in[3];
  const float* h1   = (const float*)d_in[4];
  const float* c1   = (const float*)d_in[5];
  const float* h2   = (const float*)d_in[6];
  const float* c2   = (const float*)d_in[7];
  const float* h3   = (const float*)d_in[8];
  const float* c3   = (const float*)d_in[9];
  const float* bias = (const float*)d_in[10];
  // lstm1: W_xh, W_ci, b_i, W_cf, b_f, b_c, W_co, b_o
  const float* W1   = (const float*)d_in[11];
  const float* Wci1 = (const float*)d_in[12]; const float* bi1 = (const float*)d_in[13];
  const float* Wcf1 = (const float*)d_in[14]; const float* bf1 = (const float*)d_in[15];
  const float* bc1  = (const float*)d_in[16];
  const float* Wco1 = (const float*)d_in[17]; const float* bo1 = (const float*)d_in[18];
  const float* W2   = (const float*)d_in[19];
  const float* Wci2 = (const float*)d_in[20]; const float* bi2 = (const float*)d_in[21];
  const float* Wcf2 = (const float*)d_in[22]; const float* bf2 = (const float*)d_in[23];
  const float* bc2  = (const float*)d_in[24];
  const float* Wco2 = (const float*)d_in[25]; const float* bo2 = (const float*)d_in[26];
  const float* W3   = (const float*)d_in[27];
  const float* Wci3 = (const float*)d_in[28]; const float* bi3 = (const float*)d_in[29];
  const float* Wcf3 = (const float*)d_in[30]; const float* bf3 = (const float*)d_in[31];
  const float* bc3  = (const float*)d_in[32];
  const float* Wco3 = (const float*)d_in[33]; const float* bo3 = (const float*)d_in[34];
  const float* Wa = (const float*)d_in[35]; const float* ba = (const float*)d_in[36];
  const float* Wb = (const float*)d_in[37]; const float* bb = (const float*)d_in[38];
  const float* Wk = (const float*)d_in[39]; const float* bk = (const float*)d_in[40];
  const float* Wpi = (const float*)d_in[41]; const float* bpi = (const float*)d_in[42];
  const float* Wmu = (const float*)d_in[43]; const float* bmu = (const float*)d_in[44];
  const float* Wsd = (const float*)d_in[45]; const float* bsd = (const float*)d_in[46];
  const float* Wro = (const float*)d_in[47]; const float* bro = (const float*)d_in[48];
  const float* Weo = (const float*)d_in[49]; const float* beo = (const float*)d_in[50];

  float* out = (float*)d_out;
  float*          Zbuf = (float*)d_ws;                                      // 26,214,400 B
  unsigned short* Abuf = (unsigned short*)((char*)d_ws + 26214400);         //  7,340,032 B
  unsigned short* Wbuf = (unsigned short*)((char*)d_ws + 33554432);         //  2,867,200 B (W^T)

  const dim3 gGemm(BATCH / 64, NCOL / 64);

  // ---- LSTM 1 : K = 483 -> kpad 512 ----
  convert_w_t<<<dim3(512 / 32, NCOL / 32), 256, 0, stream>>>(W1, Wbuf, 483, 512);
  gather_a<<<(BATCH * 512) / 256, 256, 0, stream>>>(Abuf, 512, x, w0, h1, (const float*)nullptr,
                                                    AWIN, HID, 0);
  gemm_bf16_wmma<<<gGemm, 128, 0, stream>>>(Abuf, Wbuf, Zbuf, 512);
  lstm_cell<<<(BATCH * HID) / 256, 256, 0, stream>>>(Zbuf, c1, Wci1, bi1, Wcf1, bf1, bc1, Wco1, bo1,
                                                     out + OFF_H1, out + OFF_C1);
  // ---- window + w_new ----
  window_k<<<BATCH, 128, 0, stream>>>(out + OFF_H1, Wa, ba, Wb, bb, Wk, bk, kin,
                                      out + OFF_K, out + OFF_PHI);
  wnew_k<<<BATCH, 128, 0, stream>>>(out + OFF_PHI, cmat, out + OFF_W);

  // ---- LSTM 2 : K = 883 -> kpad 896 ----
  convert_w_t<<<dim3(896 / 32, NCOL / 32), 256, 0, stream>>>(W2, Wbuf, 883, 896);
  gather_a<<<(BATCH * 896) / 256, 256, 0, stream>>>(Abuf, 896, x, out + OFF_H1, out + OFF_W, h2,
                                                    HID, AWIN, HID);
  gemm_bf16_wmma<<<gGemm, 128, 0, stream>>>(Abuf, Wbuf, Zbuf, 896);
  lstm_cell<<<(BATCH * HID) / 256, 256, 0, stream>>>(Zbuf, c2, Wci2, bi2, Wcf2, bf2, bc2, Wco2, bo2,
                                                     out + OFF_H2, out + OFF_C2);
  // ---- LSTM 3 ----
  convert_w_t<<<dim3(896 / 32, NCOL / 32), 256, 0, stream>>>(W3, Wbuf, 883, 896);
  gather_a<<<(BATCH * 896) / 256, 256, 0, stream>>>(Abuf, 896, x, out + OFF_H2, out + OFF_W, h3,
                                                    HID, AWIN, HID);
  gemm_bf16_wmma<<<gGemm, 128, 0, stream>>>(Abuf, Wbuf, Zbuf, 896);
  lstm_cell<<<(BATCH * HID) / 256, 256, 0, stream>>>(Zbuf, c3, Wci3, bi3, Wcf3, bf3, bc3, Wco3, bo3,
                                                     out + OFF_H3, out + OFF_C3);
  // ---- mixture heads (batch row 0 only) ----
  mix_head<<<1, 256, 0, stream>>>(out + OFF_H1, out + OFF_H2, out + OFF_H3,
                                  Wpi, bpi, Wmu, bmu, Wsd, bsd, Wro, bro, Weo, beo,
                                  bias, out);
}
